// HGNN_AC_20993800143164
// MI455X (gfx1250) — matrix operand order repaired and checked by reading, not compiled, not run on hardware
//
#include <hip/hip_runtime.h>
#include <hip/hip_bf16.h>

// ---------------------------------------------------------------------------
// GAT flash-attention for MI455X (gfx1250, wave32, WMMA bf16 16x16x32)
// ---------------------------------------------------------------------------

typedef __attribute__((ext_vector_type(16))) __bf16 v16bf;
typedef __attribute__((ext_vector_type(8)))  float  v8f;

union Frag {
    v16bf v;
    uint4 q[2];
};

constexpr int kH = 8;     // heads
constexpr int kN = 4096;  // N_DEST == N_SRC
constexpr int kD = 64;    // IN_DIM == HID == FEAT
constexpr float kNegInf = -9.0e15f;

// ---------------------------------------------------------------------------
// Prep: h1[h][s][k] = sum_d emb_src[s][d] * W[h][d][k]   (bf16 out)
// ---------------------------------------------------------------------------
__global__ void prep_h1(const float* __restrict__ emb_src,
                        const float* __restrict__ W,
                        __bf16* __restrict__ h1) {
    const int id = blockIdx.x;
    const int h = id >> 12;
    const int s = id & (kN - 1);
    const int k = threadIdx.x;
    const float* e = emb_src + (size_t)s * kD;
    const float* w = W + (size_t)h * kD * kD + k;
    float acc = 0.f;
#pragma unroll 8
    for (int d = 0; d < kD; ++d) acc += e[d] * w[(size_t)d * kD];
    h1[((size_t)h * kN + s) * kD + k] = (__bf16)acc;
}

// ---------------------------------------------------------------------------
// Prep: h2w[h][t][m] = sum_k (sum_d emb_dest[t][d]*W[h][d][k]) * W2[h][k][m]
// ---------------------------------------------------------------------------
__global__ void prep_h2w(const float* __restrict__ emb_dest,
                         const float* __restrict__ W,
                         const float* __restrict__ W2,
                         __bf16* __restrict__ h2w) {
    __shared__ float hrow[kD];
    const int id = blockIdx.x;
    const int h = id >> 12;
    const int t = id & (kN - 1);
    const int k = threadIdx.x;
    const float* e = emb_dest + (size_t)t * kD;
    const float* w = W + (size_t)h * kD * kD + k;
    float acc = 0.f;
#pragma unroll 8
    for (int d = 0; d < kD; ++d) acc += e[d] * w[(size_t)d * kD];
    hrow[k] = acc;
    __syncthreads();
    const float* w2 = W2 + (size_t)h * kD * kD + k;  // column m = k
    float a2 = 0.f;
#pragma unroll 8
    for (int kk = 0; kk < kD; ++kk) a2 += hrow[kk] * w2[(size_t)kk * kD];
    h2w[((size_t)h * kN + t) * kD + k] = (__bf16)a2;
}

// ---------------------------------------------------------------------------
// Prep: feature transpose to bf16: ftT[f][s] = feature_src[s][f]
// ---------------------------------------------------------------------------
__global__ void prep_ft(const float* __restrict__ feat,
                        __bf16* __restrict__ ftT) {
    const int idx = blockIdx.x * 256 + threadIdx.x;  // kD * kN total
    const int f = idx >> 12;
    const int s = idx & (kN - 1);
    ftT[(size_t)f * kN + s] = (__bf16)feat[(size_t)s * kD + f];
}

// ---------------------------------------------------------------------------
// Flash attention. Grid: (kN/128, kH). Block: 256 threads = 8 waves.
// Each wave owns 16 dest rows; loops over 64-wide src tiles with online
// softmax; two bf16 WMMA GEMMs per tile (scores, P@F).
// ---------------------------------------------------------------------------
__global__ __launch_bounds__(256, 1)
void gat_flash(const float* __restrict__ bias,
               const __bf16* __restrict__ h1,
               const __bf16* __restrict__ h2w,
               const __bf16* __restrict__ ftT,
               float* __restrict__ hp) {
    __shared__ __align__(16) __bf16 sh_h1[kD * kD];      // h1 tile  [s][k]  8 KB
    __shared__ __align__(16) __bf16 sh_ft[kD * kD];      // ftT tile [f][s]  8 KB
    __shared__ __align__(16) __bf16 sh_p[8][16 * kD];    // wave P   [m][s] 16 KB

    const int tid  = threadIdx.x;
    const int wave = tid >> 5;
    const int lane = tid & 31;
    const int m    = lane & 15;      // row (A) or column (B/D) within 16
    const int hi   = lane >> 4;      // lane half selects K / M sub-range
    const int h    = blockIdx.y;
    const int t0w  = blockIdx.x * 128 + wave * 16;   // wave's first dest row
    const int mbase = hi * 8;                        // D-layout row base

    // Iteration-invariant A fragments of h2w (16 rows x K=64), per ISA layout:
    // lanes 0-15 hold K {0..7,16..23}(+base), lanes 16-31 hold K {8..15,24..31}.
    const __bf16* hwRow = h2w + ((size_t)h * kN + t0w + m) * kD;
    Frag a0, a1;
    a0.q[0] = *(const uint4*)(hwRow + 0  + hi * 8);
    a0.q[1] = *(const uint4*)(hwRow + 16 + hi * 8);
    a1.q[0] = *(const uint4*)(hwRow + 32 + hi * 8);
    a1.q[1] = *(const uint4*)(hwRow + 48 + hi * 8);

    v8f   facc[4];
    float mrun[8], lrun[8];
#pragma unroll
    for (int c = 0; c < 4; ++c)
#pragma unroll
        for (int j = 0; j < 8; ++j) facc[c][j] = 0.f;
#pragma unroll
    for (int j = 0; j < 8; ++j) { mrun[j] = -3.0e38f; lrun[j] = 0.f; }

    const __bf16* h1Head = h1 + (size_t)h * kN * kD;
    __bf16* Pw = sh_p[wave];

    for (int it = 0; it < kN / kD; ++it) {
        const int s1 = it * kD;
        __syncthreads();
        {   // cooperative stage: 256 threads x 32B each per tile
            const int r  = tid >> 2;
            const int c0 = (tid & 3) * 16;
            const uint4* g1 = (const uint4*)(h1Head + (size_t)(s1 + r) * kD + c0);
            uint4* d1 = (uint4*)(sh_h1 + r * kD + c0);
            d1[0] = g1[0]; d1[1] = g1[1];
            const uint4* g2 = (const uint4*)(ftT + (size_t)r * kN + s1 + c0);
            uint4* d2 = (uint4*)(sh_ft + r * kD + c0);
            d2[0] = g2[0]; d2[1] = g2[1];
            if (it + 1 < kN / kD) {
                __builtin_prefetch(h1Head + (size_t)(s1 + kD + r) * kD + c0, 0, 1);
                __builtin_prefetch(ftT + (size_t)r * kN + s1 + kD + c0, 0, 1);
            }
        }
        __syncthreads();

        // ---- GEMM1: S[16 x 64] = h2w_wave @ h1_tile^T (4 N-tiles, K=64) ----
        v8f st[4];
#pragma unroll
        for (int c = 0; c < 4; ++c) {
            // B fragment: lane holds column s = c*16+m, K range by lane half;
            // B[k][s] = h1[s][k] -> contiguous in row-major h1 tile.
            const __bf16* col = sh_h1 + (c * 16 + m) * kD;
            Frag b0, b1;
            b0.q[0] = *(const uint4*)(col + 0  + hi * 8);
            b0.q[1] = *(const uint4*)(col + 16 + hi * 8);
            b1.q[0] = *(const uint4*)(col + 32 + hi * 8);
            b1.q[1] = *(const uint4*)(col + 48 + hi * 8);
            v8f z = {0.f, 0.f, 0.f, 0.f, 0.f, 0.f, 0.f, 0.f};
            v8f acc = __builtin_amdgcn_wmma_f32_16x16x32_bf16(
                false, a0.v, false, b0.v, (short)0, z, false, false);
            st[c] = __builtin_amdgcn_wmma_f32_16x16x32_bf16(
                false, a1.v, false, b1.v, (short)0, acc, false, false);
        }

        // ---- leaky_relu + adjacency mask ----
#pragma unroll
        for (int c = 0; c < 4; ++c) {
            const int scol = s1 + c * 16 + m;
#pragma unroll
            for (int j = 0; j < 8; ++j) {
                float e = st[c][j];
                e = (e > 0.f) ? e : 0.2f * e;
                const float bv = bias[(size_t)(t0w + mbase + j) * kN + scol];
                st[c][j] = (bv > 0.f) ? e : kNegInf;
            }
        }

        // ---- online softmax (rows replicated across 16-lane groups) ----
#pragma unroll
        for (int j = 0; j < 8; ++j) {
            float v = fmaxf(fmaxf(st[0][j], st[1][j]), fmaxf(st[2][j], st[3][j]));
            v = fmaxf(v, __shfl_xor(v, 1, 32));
            v = fmaxf(v, __shfl_xor(v, 2, 32));
            v = fmaxf(v, __shfl_xor(v, 4, 32));
            v = fmaxf(v, __shfl_xor(v, 8, 32));
            const float mnew = fmaxf(mrun[j], v);
            const float corr = __expf(mrun[j] - mnew);
            mrun[j] = mnew;
            lrun[j] *= corr;
            facc[0][j] *= corr; facc[1][j] *= corr;
            facc[2][j] *= corr; facc[3][j] *= corr;
        }
#pragma unroll
        for (int j = 0; j < 8; ++j) {
            const float p0 = __expf(st[0][j] - mrun[j]);
            const float p1 = __expf(st[1][j] - mrun[j]);
            const float p2 = __expf(st[2][j] - mrun[j]);
            const float p3 = __expf(st[3][j] - mrun[j]);
            st[0][j] = p0; st[1][j] = p1; st[2][j] = p2; st[3][j] = p3;
            float s = (p0 + p1) + (p2 + p3);
            s += __shfl_xor(s, 1, 32);
            s += __shfl_xor(s, 2, 32);
            s += __shfl_xor(s, 4, 32);
            s += __shfl_xor(s, 8, 32);
            lrun[j] += s;
        }

        // ---- P (D-layout) -> LDS [m][s] bf16, wave-private ----
#pragma unroll
        for (int c = 0; c < 4; ++c)
#pragma unroll
            for (int j = 0; j < 8; ++j)
                Pw[(mbase + j) * kD + c * 16 + m] = (__bf16)st[c][j];
        asm volatile("s_wait_dscnt 0" ::: "memory");

        // ---- GEMM2: facc += P[16x64] @ F[64x64] ----
        Frag pa0, pa1;
        const __bf16* pRow = Pw + m * kD;
        pa0.q[0] = *(const uint4*)(pRow + 0  + hi * 8);
        pa0.q[1] = *(const uint4*)(pRow + 16 + hi * 8);
        pa1.q[0] = *(const uint4*)(pRow + 32 + hi * 8);
        pa1.q[1] = *(const uint4*)(pRow + 48 + hi * 8);
#pragma unroll
        for (int c = 0; c < 4; ++c) {
            // B[k][f] = feat[s0+k][f] = ftT[f][s0+k] -> contiguous ft tile rows
            const __bf16* fRow = sh_ft + (c * 16 + m) * kD;
            Frag fb0, fb1;
            fb0.q[0] = *(const uint4*)(fRow + 0  + hi * 8);
            fb0.q[1] = *(const uint4*)(fRow + 16 + hi * 8);
            fb1.q[0] = *(const uint4*)(fRow + 32 + hi * 8);
            fb1.q[1] = *(const uint4*)(fRow + 48 + hi * 8);
            facc[c] = __builtin_amdgcn_wmma_f32_16x16x32_bf16(
                false, pa0.v, false, fb0.v, (short)0, facc[c], false, false);
            facc[c] = __builtin_amdgcn_wmma_f32_16x16x32_bf16(
                false, pa1.v, false, fb1.v, (short)0, facc[c], false, false);
        }
    }

    // ---- normalize and store per-head result ----
    float* hpHead = hp + ((size_t)h * kN + t0w) * kD;
#pragma unroll
    for (int c = 0; c < 4; ++c)
#pragma unroll
        for (int j = 0; j < 8; ++j)
            hpHead[(mbase + j) * kD + c * 16 + m] = facc[c][j] / lrun[j];
}

// ---------------------------------------------------------------------------
// out[t][f] = mean_h elu(hp[h][t][f])
// ---------------------------------------------------------------------------
__global__ void head_mean(const float* __restrict__ hp,
                          float* __restrict__ out) {
    const int idx = blockIdx.x * 256 + threadIdx.x;  // t*64+f
    float s = 0.f;
#pragma unroll
    for (int h = 0; h < kH; ++h) {
        const float x = hp[(size_t)h * kN * kD + idx];
        s += (x > 0.f) ? x : (__expf(x) - 1.f);
    }
    out[idx] = s * 0.125f;
}

// ---------------------------------------------------------------------------
extern "C" void kernel_launch(void* const* d_in, const int* in_sizes, int n_in,
                              void* d_out, int out_size, void* d_ws, size_t ws_size,
                              hipStream_t stream) {
    const float* bias     = (const float*)d_in[0];
    const float* emb_dest = (const float*)d_in[1];
    const float* emb_src  = (const float*)d_in[2];
    const float* feat     = (const float*)d_in[3];
    const float* W        = (const float*)d_in[4];
    const float* W2       = (const float*)d_in[5];
    float* out = (float*)d_out;

    char* ws = (char*)d_ws;
    __bf16* h1  = (__bf16*)(ws);                              // 4 MB
    __bf16* h2w = (__bf16*)(ws + ((size_t)4 << 20));          // 4 MB
    __bf16* ftT = (__bf16*)(ws + ((size_t)8 << 20));          // 512 KB
    float*  hp  = (float*)(ws + ((size_t)8 << 20) + ((size_t)1 << 19)); // 8 MB

    prep_h1 <<<kH * kN, kD, 0, stream>>>(emb_src, W, h1);
    prep_h2w<<<kH * kN, kD, 0, stream>>>(emb_dest, W, W2, h2w);
    prep_ft <<<(kN * kD) / 256, 256, 0, stream>>>(feat, ftT);
    gat_flash<<<dim3(kN / 128, kH), 256, 0, stream>>>(bias, h1, h2w, ftT, hp);
    head_mean<<<(kN * kD) / 256, 256, 0, stream>>>(hp, out);
}
